// LightweightGAT_36455682408740
// MI455X (gfx1250) — matrix operand order repaired and checked by reading, not compiled, not run on hardware
//
#include <hip/hip_runtime.h>

typedef __attribute__((ext_vector_type(16))) __bf16 v16bf;
typedef __attribute__((ext_vector_type(8)))  __bf16 v8bf;
typedef __attribute__((ext_vector_type(8)))  float  v8f;

#define B_   4
#define N_   2048
#define IN_  64
#define HID_ 64
#define H_   4
#define OUT_ 32
#define NW_  (N_/32)          // packed mask words per row
#define ALPHA_  0.2f
#define EMAX_   60.0f         // overflow guard; never engaged for this data

// ---------------------------------------------------------------------------
// K0: pack adj[B,N,N] (int 0/1) into bitmasks, CHUNK-MAJOR: mpk[b][c][i],
// word = bits j = 32c..32c+31 of row i. One wave per row; wave32 ballot
// produces the word directly. adj (67 MB) is read exactly once.
// ---------------------------------------------------------------------------
__global__ __launch_bounds__(256) void gat55_pack_adj(const int* __restrict__ adj,
                                                      unsigned* __restrict__ mpk) {
    const int lane = threadIdx.x & 31;
    const int wave = threadIdx.x >> 5;
    const int row  = blockIdx.x * 8 + wave;          // flattened (b*N + i)
    const int b = row >> 11;                         // /N_
    const int i = row & (N_ - 1);
    const int* arow = adj + (size_t)row * N_;
    unsigned* mcol  = mpk + (size_t)b * NW_ * N_ + i;
    for (int c = 0; c < NW_; ++c) {
        int v = arow[c * 32 + lane];
        unsigned m = __builtin_amdgcn_ballot_w32(v > 0);
        if (lane == 0) mcol[(size_t)c * N_] = m;
    }
}

// ---------------------------------------------------------------------------
// K1: layer-1 feature transform (f32): Wh = x @ W1[h]^T per head.
// Emits Wh as bf16 in f-major layout whb1[b][h][f][n] (WMMA B-operand
// friendly) plus s_src = Wh.a1[:64], s_dst = Wh.a1[64:].
// ---------------------------------------------------------------------------
__global__ __launch_bounds__(64) void gat55_l1_feat(const float* __restrict__ x,
                                                    const float* __restrict__ W1,
                                                    const float* __restrict__ a1,
                                                    __bf16* __restrict__ whb1,
                                                    float* __restrict__ s_src,
                                                    float* __restrict__ s_dst) {
    int idx = blockIdx.x;
    const int n  = idx % N_;  idx /= N_;
    const int h  = idx % H_;
    const int b  = idx / H_;
    const int t  = threadIdx.x;          // output feature f (0..63)

    __shared__ float xs[IN_];
    __shared__ float red[4];

    xs[t] = x[((size_t)b * N_ + n) * IN_ + t];
    __syncthreads();

    const float* wrow = W1 + ((size_t)h * HID_ + t) * IN_;
    float acc = 0.f;
    #pragma unroll 8
    for (int i = 0; i < IN_; ++i) acc += xs[i] * wrow[i];

    whb1[(((size_t)b * H_ + h) * HID_ + t) * N_ + n] = (__bf16)acc;

    float ps = acc * a1[h * 2 * HID_ + t];
    float pd = acc * a1[h * 2 * HID_ + HID_ + t];
    for (int off = 16; off; off >>= 1) {
        ps += __shfl_xor(ps, off, 32);
        pd += __shfl_xor(pd, off, 32);
    }
    const int lane = t & 31, wv = t >> 5;
    if (lane == 0) { red[wv] = ps; red[2 + wv] = pd; }
    __syncthreads();
    if (t == 0) {
        s_src[((size_t)b * H_ + h) * N_ + n] = red[0] + red[1];
        s_dst[((size_t)b * H_ + h) * N_ + n] = red[2] + red[3];
    }
}

// ---------------------------------------------------------------------------
// K2: layer-1 attention: single-pass un-shifted softmax + bf16 WMMA + ELU.
// p = exp(clamp(e,EMAX)) (masked -> 0); normalize by row-sum at the end —
// mathematically identical to max-shifted softmax, no stats scan needed.
// 128 threads (4 waves) per 16-row tile; wave w owns output features
// [16w,16w+16) and probability rows {4w..4w+3}. Double-buffered LDS p-tile,
// 2 WMMAs per barrier.
// ---------------------------------------------------------------------------
__global__ __launch_bounds__(128) void gat55_l1_attn(const float* __restrict__ s_src,
                                                     const float* __restrict__ s_dst,
                                                     const unsigned* __restrict__ mpk,
                                                     const __bf16* __restrict__ whb,
                                                     float* __restrict__ h1) {
    const int t    = threadIdx.x;
    const int lane = t & 31;
    const int wv   = t >> 5;             // 0..3
    const int half = lane >> 4;          // WMMA lane-half
    const int col  = lane & 15;          // WMMA N index

    int idx = blockIdx.x;
    const int ib = idx % (N_ / 16); idx /= (N_ / 16);
    const int h  = idx % H_;
    const int b  = idx / H_;
    const int i0 = ib * 16;

    __shared__ float ssrc[16];
    __shared__ float l_sh[16];
    __shared__ __align__(32) __bf16 ptile[2][2][16][32];   // [buf][subchunk][row][j]

    const float*    sdp   = s_dst + ((size_t)b * H_ + h) * N_;
    const unsigned* mbase = mpk + (size_t)b * NW_ * N_ + i0;   // + c*N_ + row
    const __bf16*   whead = whb + ((size_t)b * H_ + h) * HID_ * N_;

    if (t < 16) ssrc[t] = s_src[((size_t)b * H_ + h) * N_ + i0 + t];
    __syncthreads();

    float sl[4] = {0.f, 0.f, 0.f, 0.f};
    v8f acc = {};
    const int fbase = wv * 16;
    int cur = 0;

    for (int jb = 0; jb < N_; jb += 64) {
        #pragma unroll
        for (int s = 0; s < 2; ++s) {
            const int j0 = jb + 32 * s;
            float sd = sdp[j0 + lane];
            const unsigned* mch = mbase + (size_t)(j0 >> 5) * N_;
            #pragma unroll
            for (int r = 0; r < 4; ++r) {
                const int row = wv * 4 + r;
                unsigned mw = mch[row];                      // 4 contiguous words
                float e = ssrc[row] + sd;
                e = (e > 0.f) ? e : ALPHA_ * e;              // leaky relu
                e = fminf(e, EMAX_);                         // overflow guard
                float p = __expf(e);
                p = ((mw >> lane) & 1u) ? p : 0.f;           // mask -> exactly 0
                sl[r] += p;
                ptile[cur][s][row][lane] = (__bf16)p;
            }
        }
        __syncthreads();

        // A operands: 16x32 bf16 layout -> 16 contiguous K elems/lane (k = 16*half + s)
        const int arow = lane & 15;
        union { v16bf v; v8bf h8[2]; } A0, A1;
        A0.h8[0] = *(const v8bf*)&ptile[cur][0][arow][half * 16];
        A0.h8[1] = *(const v8bf*)&ptile[cur][0][arow][half * 16 + 8];
        A1.h8[0] = *(const v8bf*)&ptile[cur][1][arow][half * 16];
        A1.h8[1] = *(const v8bf*)&ptile[cur][1][arow][half * 16 + 8];
        // B operands: column f = fbase+col, K contiguous along N (f-major Wh)
        const __bf16* bp = whead + (size_t)(fbase + col) * N_ + jb + half * 16;
        v16bf Bm0 = *(const v16bf*)bp;
        v16bf Bm1 = *(const v16bf*)(bp + 32);

        acc = __builtin_amdgcn_wmma_f32_16x16x32_bf16(false, A0.v, false, Bm0,
                                                      (short)0, acc, false, false);
        acc = __builtin_amdgcn_wmma_f32_16x16x32_bf16(false, A1.v, false, Bm1,
                                                      (short)0, acc, false, false);
        cur ^= 1;
    }

    // ---- row sums -> LDS, normalize, ELU, store ----
    #pragma unroll
    for (int r = 0; r < 4; ++r)
        for (int off = 16; off; off >>= 1)
            sl[r] += __shfl_xor(sl[r], off, 32);
    if (lane == 0) {
        #pragma unroll
        for (int r = 0; r < 4; ++r) l_sh[wv * 4 + r] = sl[r];
    }
    __syncthreads();

    float* orow = h1 + ((size_t)b * N_ + i0) * (H_ * HID_) + h * HID_ + fbase + col;
    #pragma unroll
    for (int r = 0; r < 8; ++r) {
        const int row = r + 8 * half;                       // C layout: vgpr r -> row
        float v = acc[r] / fmaxf(l_sh[row], 1e-30f);
        v = (v > 0.f) ? v : (__expf(v) - 1.f);              // ELU (concat head)
        orow[(size_t)row * (H_ * HID_)] = v;
    }
}

// ---------------------------------------------------------------------------
// K3: layer-2 feature transform: Wh2 = h1 @ W2^T (f32), + s2 scalars,
// bf16 f-major copy whb2[b][f][n]. One wave per (b,n); thread t = f (0..31).
// ---------------------------------------------------------------------------
__global__ __launch_bounds__(32) void gat55_l2_feat(const float* __restrict__ h1,
                                                    const float* __restrict__ W2,
                                                    const float* __restrict__ a2,
                                                    __bf16* __restrict__ whb2,
                                                    float* __restrict__ s2_src,
                                                    float* __restrict__ s2_dst) {
    const int idx = blockIdx.x;          // b*N + n
    const int t   = threadIdx.x;         // f (0..31)
    __shared__ float hs[H_ * HID_];
    const float* hrow = h1 + (size_t)idx * (H_ * HID_);
    #pragma unroll
    for (int k = 0; k < 8; ++k) hs[t + 32 * k] = hrow[t + 32 * k];
    __syncthreads();

    const float* wrow = W2 + (size_t)t * (H_ * HID_);
    float acc = 0.f;
    #pragma unroll 8
    for (int i = 0; i < H_ * HID_; ++i) acc += hs[i] * wrow[i];

    whb2[((size_t)(idx / N_) * OUT_ + t) * N_ + (idx % N_)] = (__bf16)acc;

    float ps = acc * a2[t], pd = acc * a2[OUT_ + t];
    for (int off = 16; off; off >>= 1) {
        ps += __shfl_xor(ps, off, 32);
        pd += __shfl_xor(pd, off, 32);
    }
    if (t == 0) { s2_src[idx] = ps; s2_dst[idx] = pd; }
}

// ---------------------------------------------------------------------------
// K4: layer-2 attention (single head, Fout=32, no ELU). 2 waves / block:
// wave w owns features [16w,16w+16) and probability rows {8w..8w+7}.
// Same single-pass + double-buffer structure as K2.
// ---------------------------------------------------------------------------
__global__ __launch_bounds__(64) void gat55_l2_attn(const float* __restrict__ s_src,
                                                    const float* __restrict__ s_dst,
                                                    const unsigned* __restrict__ mpk,
                                                    const __bf16* __restrict__ whb,
                                                    float* __restrict__ out) {
    const int t    = threadIdx.x;
    const int lane = t & 31;
    const int wv   = t >> 5;             // 0..1
    const int half = lane >> 4;
    const int col  = lane & 15;

    int idx = blockIdx.x;
    const int ib = idx % (N_ / 16);
    const int b  = idx / (N_ / 16);
    const int i0 = ib * 16;

    __shared__ float ssrc[16];
    __shared__ float l_sh[16];
    __shared__ __align__(32) __bf16 ptile[2][2][16][32];

    const float*    sdp   = s_dst + (size_t)b * N_;
    const unsigned* mbase = mpk + (size_t)b * NW_ * N_ + i0;
    const __bf16*   whead = whb + (size_t)b * OUT_ * N_;

    if (t < 16) ssrc[t] = s_src[(size_t)b * N_ + i0 + t];
    __syncthreads();

    float sl[8] = {0.f, 0.f, 0.f, 0.f, 0.f, 0.f, 0.f, 0.f};
    v8f acc = {};
    const int fbase = wv * 16;
    int cur = 0;

    for (int jb = 0; jb < N_; jb += 64) {
        #pragma unroll
        for (int s = 0; s < 2; ++s) {
            const int j0 = jb + 32 * s;
            float sd = sdp[j0 + lane];
            const unsigned* mch = mbase + (size_t)(j0 >> 5) * N_;
            #pragma unroll
            for (int r = 0; r < 8; ++r) {
                const int row = wv * 8 + r;
                unsigned mw = mch[row];
                float e = ssrc[row] + sd;
                e = (e > 0.f) ? e : ALPHA_ * e;
                e = fminf(e, EMAX_);
                float p = __expf(e);
                p = ((mw >> lane) & 1u) ? p : 0.f;
                sl[r] += p;
                ptile[cur][s][row][lane] = (__bf16)p;
            }
        }
        __syncthreads();

        const int arow = lane & 15;
        union { v16bf v; v8bf h8[2]; } A0, A1;
        A0.h8[0] = *(const v8bf*)&ptile[cur][0][arow][half * 16];
        A0.h8[1] = *(const v8bf*)&ptile[cur][0][arow][half * 16 + 8];
        A1.h8[0] = *(const v8bf*)&ptile[cur][1][arow][half * 16];
        A1.h8[1] = *(const v8bf*)&ptile[cur][1][arow][half * 16 + 8];
        const __bf16* bp = whead + (size_t)(fbase + col) * N_ + jb + half * 16;
        v16bf Bm0 = *(const v16bf*)bp;
        v16bf Bm1 = *(const v16bf*)(bp + 32);

        acc = __builtin_amdgcn_wmma_f32_16x16x32_bf16(false, A0.v, false, Bm0,
                                                      (short)0, acc, false, false);
        acc = __builtin_amdgcn_wmma_f32_16x16x32_bf16(false, A1.v, false, Bm1,
                                                      (short)0, acc, false, false);
        cur ^= 1;
    }

    #pragma unroll
    for (int r = 0; r < 8; ++r)
        for (int off = 16; off; off >>= 1)
            sl[r] += __shfl_xor(sl[r], off, 32);
    if (lane == 0) {
        #pragma unroll
        for (int r = 0; r < 8; ++r) l_sh[wv * 8 + r] = sl[r];
    }
    __syncthreads();

    float* orow = out + ((size_t)b * N_ + i0) * OUT_ + fbase + col;
    #pragma unroll
    for (int r = 0; r < 8; ++r) {
        const int row = r + 8 * half;
        orow[(size_t)row * OUT_] = acc[r] / fmaxf(l_sh[row], 1e-30f);   // no ELU
    }
}

// ---------------------------------------------------------------------------
extern "C" void kernel_launch(void* const* d_in, const int* in_sizes, int n_in,
                              void* d_out, int out_size, void* d_ws, size_t ws_size,
                              hipStream_t stream) {
    const float* x  = (const float*)d_in[0];
    const int*   adj= (const int*)d_in[1];
    const float* W1 = (const float*)d_in[2];
    const float* a1 = (const float*)d_in[3];
    const float* W2 = (const float*)d_in[4];
    const float* a2 = (const float*)d_in[5];
    float* out = (float*)d_out;

    char* ws = (char*)d_ws;
    size_t off = 0;
    auto carve = [&](size_t bytes) {
        size_t o = off;
        off = (off + bytes + 255) & ~(size_t)255;
        return o;
    };
    unsigned* mpk  = (unsigned*)(ws + carve((size_t)B_ * N_ * NW_ * sizeof(unsigned)));
    __bf16*   whb1 = (__bf16*)  (ws + carve((size_t)B_ * H_ * HID_ * N_ * sizeof(__bf16)));
    float*    s1s  = (float*)   (ws + carve((size_t)B_ * H_ * N_ * sizeof(float)));
    float*    s1d  = (float*)   (ws + carve((size_t)B_ * H_ * N_ * sizeof(float)));
    float*    h1   = (float*)   (ws + carve((size_t)B_ * N_ * H_ * HID_ * sizeof(float)));
    __bf16*   whb2 = (__bf16*)  (ws + carve((size_t)B_ * OUT_ * N_ * sizeof(__bf16)));
    float*    s2s  = (float*)   (ws + carve((size_t)B_ * N_ * sizeof(float)));
    float*    s2d  = (float*)   (ws + carve((size_t)B_ * N_ * sizeof(float)));

    gat55_pack_adj<<<dim3(B_ * N_ / 8), dim3(256), 0, stream>>>(adj, mpk);
    gat55_l1_feat <<<dim3(B_ * H_ * N_), dim3(64), 0, stream>>>(x, W1, a1, whb1, s1s, s1d);
    gat55_l1_attn <<<dim3(B_ * H_ * (N_ / 16)), dim3(128), 0, stream>>>(s1s, s1d, mpk, whb1, h1);
    gat55_l2_feat <<<dim3(B_ * N_), dim3(32), 0, stream>>>(h1, W2, a2, whb2, s2s, s2d);
    gat55_l2_attn <<<dim3(B_ * (N_ / 16)), dim3(64), 0, stream>>>(s2s, s2d, mpk, whb2, out);
}
